// GCN_Net_84877143703894
// MI455X (gfx1250) — matrix operand order repaired and checked by reading, not compiled, not run on hardware
//
#include <hip/hip_runtime.h>
#include <math.h>

#define N_NODES 50000
#define N_EDGES 800000
#define F_IN    64
#define F_HID   128
#define N_CLS   40
#define N_CLS_P 48   // padded to 3 WMMA n-tiles

typedef __attribute__((ext_vector_type(16))) _Float16 v16h;
typedef __attribute__((ext_vector_type(8)))  float    v8f;

// ---------------- degree / normalization ----------------

__global__ void k_init_deg(float* __restrict__ deg) {
    int i = blockIdx.x * blockDim.x + threadIdx.x;
    if (i < N_NODES) deg[i] = 1.0f;              // self loop
}

__global__ void k_deg_edges(const int* __restrict__ ei, float* __restrict__ deg) {
    int e = blockIdx.x * blockDim.x + threadIdx.x;
    if (e < N_EDGES) atomicAdd(&deg[ei[N_EDGES + e]], 1.0f);
}

__global__ void k_dinv(const float* __restrict__ deg, float* __restrict__ dinv) {
    int i = blockIdx.x * blockDim.x + threadIdx.x;
    if (i < N_NODES) dinv[i] = rsqrtf(deg[i]);
}

// ---------------- GEMM1: H1[N,128] = x[N,64] @ W1[64,128] (f16 WMMA) --------
// LDS holds W1 pre-permuted into B-fragment layout: fragment (kc,nt) is 32
// lanes x 16 f16, contiguous, so each lane reads its whole fragment slice as
// one 32B vector LDS load.

__global__ __launch_bounds__(256) void k_gemm1(const float* __restrict__ x,
                                               const float* __restrict__ W,
                                               float* __restrict__ out) {
    __shared__ alignas(32) _Float16 ws[2 * 8 * 32 * 16];   // 16 KB, fragment order
    for (int g = threadIdx.x; g < 2 * 8 * 32; g += 256) {
        const int kc   = g >> 8;          // fragment k-chunk
        const int rem  = g & 255;
        const int nt   = rem >> 5;        // fragment n-tile
        const int l    = rem & 31;        // lane slot
        const int half = l >> 4;
        const int lr   = l & 15;
        const int col  = nt * 16 + lr;
        const int krow = kc * 32 + half * 8;
        v16h tmp;
        #pragma unroll
        for (int j = 0; j < 16; ++j)
            tmp[j] = (_Float16)W[(krow + j + (j >= 8 ? 8 : 0)) * F_HID + col];
        *(v16h*)&ws[(size_t)g * 16] = tmp;
    }
    __syncthreads();

    const int wave = threadIdx.x >> 5;
    const int lane = threadIdx.x & 31;
    const int half = lane >> 4;
    const int lr   = lane & 15;
    const int rowBase = blockIdx.x * 128 + wave * 16;

    const v8f vzero = {0.f,0.f,0.f,0.f,0.f,0.f,0.f,0.f};
    v8f acc[8];
    #pragma unroll
    for (int t = 0; t < 8; ++t) acc[t] = vzero;

    const int  arow   = rowBase + lr;
    const bool avalid = arow < N_NODES;
    const float* xrow = x + (size_t)arow * F_IN;
    const int  koff   = half * 8;

    #pragma unroll
    for (int kc = 0; kc < F_IN / 32; ++kc) {
        const int k0 = kc * 32;
        float av[16];
        if (avalid) {
            const float4 p0 = *(const float4*)(xrow + k0 + koff);
            const float4 p1 = *(const float4*)(xrow + k0 + koff + 4);
            const float4 q0 = *(const float4*)(xrow + k0 + koff + 16);
            const float4 q1 = *(const float4*)(xrow + k0 + koff + 20);
            av[0]=p0.x; av[1]=p0.y; av[2] =p0.z; av[3] =p0.w;
            av[4]=p1.x; av[5]=p1.y; av[6] =p1.z; av[7] =p1.w;
            av[8]=q0.x; av[9]=q0.y; av[10]=q0.z; av[11]=q0.w;
            av[12]=q1.x; av[13]=q1.y; av[14]=q1.z; av[15]=q1.w;
        } else {
            #pragma unroll
            for (int j = 0; j < 16; ++j) av[j] = 0.0f;
        }
        v16h a;
        #pragma unroll
        for (int j = 0; j < 16; ++j) a[j] = (_Float16)av[j];

        #pragma unroll
        for (int nt = 0; nt < 8; ++nt) {
            const v16h b = *(const v16h*)&ws[(((kc * 8 + nt) * 32 + lane) << 4)];
            acc[nt] = __builtin_amdgcn_wmma_f32_16x16x32_f16(
                false, a, false, b, (short)0, acc[nt], false, false);
        }
    }

    #pragma unroll
    for (int nt = 0; nt < 8; ++nt) {
        #pragma unroll
        for (int r = 0; r < 8; ++r) {
            const int orow = rowBase + half * 8 + r;
            if (orow < N_NODES)
                out[(size_t)orow * F_HID + nt * 16 + lr] = acc[nt][r];
        }
    }
}

// ---------------- layer-1 aggregation -----------------------------

__global__ void k_init1(const float* __restrict__ dinv, const float* __restrict__ h,
                        float* __restrict__ acc) {
    long long idx = (long long)blockIdx.x * blockDim.x + threadIdx.x;
    if (idx < (long long)N_NODES * F_HID) {
        int i = (int)(idx >> 7);
        float s = dinv[i];
        acc[idx] = s * s * h[idx];               // self-loop contribution
    }
}

__global__ void k_edge1(const int* __restrict__ ei, const float* __restrict__ dinv,
                        const float* __restrict__ h, float* __restrict__ acc) {
    long long idx = (long long)blockIdx.x * blockDim.x + threadIdx.x;
    if (idx >= (long long)N_EDGES * 32) return;
    const int e  = (int)(idx >> 5);
    const int f4 = (int)(idx & 31) << 2;
    const int r = ei[e];
    const int c = ei[N_EDGES + e];
    const float w = dinv[r] * dinv[c];
    const float4 hv = *(const float4*)(h + (size_t)r * F_HID + f4);
    float* d = acc + (size_t)c * F_HID + f4;
    atomicAdd(d + 0, hv.x * w);
    atomicAdd(d + 1, hv.y * w);
    atomicAdd(d + 2, hv.z * w);
    atomicAdd(d + 3, hv.w * w);
}

__global__ void k_fin1(float* __restrict__ acc, const float* __restrict__ b1,
                       const float* __restrict__ alpha) {
    long long idx = (long long)blockIdx.x * blockDim.x + threadIdx.x;
    if (idx < (long long)N_NODES * F_HID) {
        int f = (int)(idx & (F_HID - 1));
        float v = acc[idx] + b1[f];
        acc[idx] = (v >= 0.0f) ? v : alpha[0] * v;   // PReLU
    }
}

// ---------------- GEMM2: H2[N,40] = X2[N,128] @ W2[128,40] (padded 48) ------

__global__ __launch_bounds__(256) void k_gemm2(const float* __restrict__ x,
                                               const float* __restrict__ W,
                                               float* __restrict__ out) {
    __shared__ alignas(32) _Float16 ws[4 * 3 * 32 * 16];   // 12 KB, fragment order
    for (int g = threadIdx.x; g < 4 * 3 * 32; g += 256) {
        const int kc   = g / 96;
        const int rem  = g % 96;
        const int nt   = rem >> 5;
        const int l    = rem & 31;
        const int half = l >> 4;
        const int lr   = l & 15;
        const int col  = nt * 16 + lr;
        const int krow = kc * 32 + half * 8;
        v16h tmp;
        #pragma unroll
        for (int j = 0; j < 16; ++j)
            tmp[j] = (col < N_CLS) ? (_Float16)W[(krow + j + (j >= 8 ? 8 : 0)) * N_CLS + col]
                                   : (_Float16)0.0f;
        *(v16h*)&ws[(size_t)g * 16] = tmp;
    }
    __syncthreads();

    const int wave = threadIdx.x >> 5;
    const int lane = threadIdx.x & 31;
    const int half = lane >> 4;
    const int lr   = lane & 15;
    const int rowBase = blockIdx.x * 128 + wave * 16;

    const v8f vzero = {0.f,0.f,0.f,0.f,0.f,0.f,0.f,0.f};
    v8f acc[3];
    #pragma unroll
    for (int t = 0; t < 3; ++t) acc[t] = vzero;

    const int  arow   = rowBase + lr;
    const bool avalid = arow < N_NODES;
    const float* xrow = x + (size_t)arow * F_HID;
    const int  koff   = half * 8;

    #pragma unroll
    for (int kc = 0; kc < F_HID / 32; ++kc) {
        const int k0 = kc * 32;
        float av[16];
        if (avalid) {
            const float4 p0 = *(const float4*)(xrow + k0 + koff);
            const float4 p1 = *(const float4*)(xrow + k0 + koff + 4);
            const float4 q0 = *(const float4*)(xrow + k0 + koff + 16);
            const float4 q1 = *(const float4*)(xrow + k0 + koff + 20);
            av[0]=p0.x; av[1]=p0.y; av[2] =p0.z; av[3] =p0.w;
            av[4]=p1.x; av[5]=p1.y; av[6] =p1.z; av[7] =p1.w;
            av[8]=q0.x; av[9]=q0.y; av[10]=q0.z; av[11]=q0.w;
            av[12]=q1.x; av[13]=q1.y; av[14]=q1.z; av[15]=q1.w;
        } else {
            #pragma unroll
            for (int j = 0; j < 16; ++j) av[j] = 0.0f;
        }
        v16h a;
        #pragma unroll
        for (int j = 0; j < 16; ++j) a[j] = (_Float16)av[j];

        #pragma unroll
        for (int nt = 0; nt < 3; ++nt) {
            const v16h b = *(const v16h*)&ws[(((kc * 3 + nt) * 32 + lane) << 4)];
            acc[nt] = __builtin_amdgcn_wmma_f32_16x16x32_f16(
                false, a, false, b, (short)0, acc[nt], false, false);
        }
    }

    #pragma unroll
    for (int nt = 0; nt < 3; ++nt) {
        #pragma unroll
        for (int r = 0; r < 8; ++r) {
            const int orow = rowBase + half * 8 + r;
            const int ocol = nt * 16 + lr;
            if (orow < N_NODES && ocol < N_CLS)
                out[(size_t)orow * N_CLS + ocol] = acc[nt][r];
        }
    }
}

// ---------------- layer-2 aggregation + log-softmax ------------------

__global__ void k_init2(const float* __restrict__ dinv, const float* __restrict__ h,
                        float* __restrict__ acc) {
    long long idx = (long long)blockIdx.x * blockDim.x + threadIdx.x;
    if (idx < (long long)N_NODES * N_CLS) {
        int i = (int)(idx / N_CLS);
        float s = dinv[i];
        acc[idx] = s * s * h[idx];
    }
}

__global__ void k_edge2(const int* __restrict__ ei, const float* __restrict__ dinv,
                        const float* __restrict__ h, float* __restrict__ acc) {
    long long idx = (long long)blockIdx.x * blockDim.x + threadIdx.x;
    if (idx >= (long long)N_EDGES * 10) return;
    const int e  = (int)(idx / 10);
    const int f4 = (int)(idx % 10) << 2;
    const int r = ei[e];
    const int c = ei[N_EDGES + e];
    const float w = dinv[r] * dinv[c];
    const float4 hv = *(const float4*)(h + (size_t)r * N_CLS + f4);
    float* d = acc + (size_t)c * N_CLS + f4;
    atomicAdd(d + 0, hv.x * w);
    atomicAdd(d + 1, hv.y * w);
    atomicAdd(d + 2, hv.z * w);
    atomicAdd(d + 3, hv.w * w);
}

// one wave32 per row: bias + log_softmax over 40 classes, in place in d_out
__global__ __launch_bounds__(256) void k_logsoftmax(float* __restrict__ out,
                                                    const float* __restrict__ b2) {
    const int lane = threadIdx.x & 31;
    const int row  = blockIdx.x * 8 + (threadIdx.x >> 5);
    if (row >= N_NODES) return;
    float* p = out + (size_t)row * N_CLS;
    float v0 = p[lane] + b2[lane];
    float v1 = (lane < 8) ? (p[32 + lane] + b2[32 + lane]) : -3.0e38f;
    float m = fmaxf(v0, v1);
    #pragma unroll
    for (int o = 16; o; o >>= 1) m = fmaxf(m, __shfl_xor(m, o, 32));
    float s = __expf(v0 - m) + ((lane < 8) ? __expf(v1 - m) : 0.0f);
    #pragma unroll
    for (int o = 16; o; o >>= 1) s += __shfl_xor(s, o, 32);
    const float l = m + __logf(s);
    p[lane] = v0 - l;
    if (lane < 8) p[32 + lane] = v1 - l;
}

// ---------------- host launcher ----------------

extern "C" void kernel_launch(void* const* d_in, const int* in_sizes, int n_in,
                              void* d_out, int out_size, void* d_ws, size_t ws_size,
                              hipStream_t stream) {
    (void)in_sizes; (void)n_in; (void)out_size; (void)ws_size;
    const float* x     = (const float*)d_in[0];
    const int*   ei    = (const int*)  d_in[1];   // [2,E] flat: row then col
    const float* W1    = (const float*)d_in[2];
    const float* b1    = (const float*)d_in[3];
    const float* W2    = (const float*)d_in[4];
    const float* b2    = (const float*)d_in[5];
    const float* alpha = (const float*)d_in[6];
    float* out = (float*)d_out;

    float* deg  = (float*)d_ws;                          // N
    float* dinv = deg  + N_NODES;                        // N
    float* bufA = dinv + N_NODES;                        // N*128  (H1, then H2)
    float* bufB = bufA + (size_t)N_NODES * F_HID;        // N*128  (acc1 -> X2)

    const int T = 256;
    // normalization
    k_init_deg <<<(N_NODES + T - 1) / T, T, 0, stream>>>(deg);
    k_deg_edges<<<(N_EDGES + T - 1) / T, T, 0, stream>>>(ei, deg);
    k_dinv     <<<(N_NODES + T - 1) / T, T, 0, stream>>>(deg, dinv);

    const int gemmGrid = (N_NODES + 127) / 128;
    // layer 1
    k_gemm1<<<gemmGrid, T, 0, stream>>>(x, W1, bufA);
    {
        long long n1 = (long long)N_NODES * F_HID;
        long long nE = (long long)N_EDGES * 32;
        k_init1<<<(unsigned)((n1 + T - 1) / T), T, 0, stream>>>(dinv, bufA, bufB);
        k_edge1<<<(unsigned)((nE + T - 1) / T), T, 0, stream>>>(ei, dinv, bufA, bufB);
        k_fin1 <<<(unsigned)((n1 + T - 1) / T), T, 0, stream>>>(bufB, b1, alpha);
    }
    // layer 2
    k_gemm2<<<gemmGrid, T, 0, stream>>>(bufB, W2, bufA);
    {
        long long n2 = (long long)N_NODES * N_CLS;
        long long nE = (long long)N_EDGES * 10;
        k_init2<<<(unsigned)((n2 + T - 1) / T), T, 0, stream>>>(dinv, bufA, out);
        k_edge2<<<(unsigned)((nE + T - 1) / T), T, 0, stream>>>(ei, dinv, bufA, out);
    }
    k_logsoftmax<<<(N_NODES + 7) / 8, T, 0, stream>>>(out, b2);
}
